// BitLinear_65687229825612
// MI455X (gfx1250) — 8-bit matrix operand order repaired and checked by reading, compiled, not run on hardware
//
#include <hip/hip_runtime.h>
#include <cstdint>


#define B_DIM   4
#define T_DIM   2048
#define D_IN    4096
#define D_OUT   16384
#define M_TOT   (B_DIM * T_DIM)
#define W_ELEMS ((size_t)D_OUT * D_IN)
#define X_ELEMS ((size_t)M_TOT * D_IN)

typedef __attribute__((ext_vector_type(8))) int v8i;
typedef int v4i_ld __attribute__((vector_size(16)));

#if __has_builtin(__builtin_amdgcn_global_load_async_to_lds_b128)
#define HAVE_ASYNC_LDS 1
#else
#define HAVE_ASYNC_LDS 0
#endif

__device__ __forceinline__ void cp16_g2l(const int8_t* g, int8_t* l) {
#if HAVE_ASYNC_LDS
    __builtin_amdgcn_global_load_async_to_lds_b128((v4i_ld*)g, (v4i_ld*)l, 0, 0);
#else
    *(int4*)l = *(const int4*)g;
#endif
}

template <int N>
__device__ __forceinline__ void wait_async() {
#if HAVE_ASYNC_LDS
#if __has_builtin(__builtin_amdgcn_s_wait_asynccnt)
    __builtin_amdgcn_s_wait_asynccnt(N);
#else
    asm volatile("s_wait_asynccnt %0" ::"i"(N) : "memory");
#endif
#endif
}

__global__ void init_ws_kernel(float* sum_abs) {
    if (threadIdx.x == 0 && blockIdx.x == 0) sum_abs[0] = 0.0f;
}

__global__ __launch_bounds__(256) void wabs_sum_kernel(const float* __restrict__ w,
                                                       float* __restrict__ sum_abs) {
    __shared__ float red[256];
    const int tid = threadIdx.x;
    const size_t n4 = W_ELEMS / 4;
    const float4* w4 = (const float4*)w;
    float s = 0.0f;
    for (size_t i = (size_t)blockIdx.x * 256 + tid; i < n4; i += (size_t)gridDim.x * 256) {
        float4 v = w4[i];
        s += fabsf(v.x) + fabsf(v.y) + fabsf(v.z) + fabsf(v.w);
    }
    red[tid] = s;
    __syncthreads();
    for (int off = 128; off > 0; off >>= 1) {
        if (tid < off) red[tid] += red[tid + off];
        __syncthreads();
    }
    if (tid == 0) atomicAdd(sum_abs, red[0]);
}

__global__ __launch_bounds__(256) void wquant_kernel(const float* __restrict__ w,
                                                     const float* __restrict__ sum_abs,
                                                     int8_t* __restrict__ wq) {
    const float wscale = sum_abs[0] * (1.0f / (float)W_ELEMS);
    const float inv = 1.0f / (wscale + 1e-8f);
    const size_t n4 = W_ELEMS / 4;
    const float4* w4 = (const float4*)w;
    int* wq4 = (int*)wq;
    for (size_t i = (size_t)blockIdx.x * 256 + threadIdx.x; i < n4;
         i += (size_t)gridDim.x * 256) {
        float4 v = w4[i];
        int q0 = (fabsf(v.x * inv) > 0.5f) ? ((v.x > 0.0f) ? 1 : -1) : 0;
        int q1 = (fabsf(v.y * inv) > 0.5f) ? ((v.y > 0.0f) ? 1 : -1) : 0;
        int q2 = (fabsf(v.z * inv) > 0.5f) ? ((v.z > 0.0f) ? 1 : -1) : 0;
        int q3 = (fabsf(v.w * inv) > 0.5f) ? ((v.w > 0.0f) ? 1 : -1) : 0;
        wq4[i] = (q0 & 0xff) | ((q1 & 0xff) << 8) | ((q2 & 0xff) << 16) | ((q3 & 0xff) << 24);
    }
}

__global__ __launch_bounds__(256) void xquant_kernel(const float* __restrict__ x,
                                                     int8_t* __restrict__ xq,
                                                     float* __restrict__ xscale) {
    __shared__ float red[256];
    const int tid = threadIdx.x;
    const int token = blockIdx.x;
    const float* row = x + (size_t)token * D_IN + tid * 16;

    float4 v[4];
    float m = 0.0f;
#pragma unroll
    for (int i = 0; i < 4; ++i) {
        v[i] = ((const float4*)row)[i];
        m = fmaxf(m, fmaxf(fmaxf(fabsf(v[i].x), fabsf(v[i].y)),
                           fmaxf(fabsf(v[i].z), fabsf(v[i].w))));
    }
    red[tid] = m;
    __syncthreads();
    for (int off = 128; off > 0; off >>= 1) {
        if (tid < off) red[tid] = fmaxf(red[tid], red[tid + off]);
        __syncthreads();
    }
    const float scale = red[0] * (1.0f / 127.0f);
    const float inv = 1.0f / (scale + 1e-8f);

    int out[4];
#pragma unroll
    for (int i = 0; i < 4; ++i) {
        float e[4] = {v[i].x, v[i].y, v[i].z, v[i].w};
        int packed = 0;
#pragma unroll
        for (int j = 0; j < 4; ++j) {
            int q = (int)rintf(e[j] * inv);
            q = min(127, max(-128, q));
            packed |= (q & 0xff) << (8 * j);
        }
        out[i] = packed;
    }
    int4* dst = (int4*)(xq + (size_t)token * D_IN + tid * 16);
    *dst = make_int4(out[0], out[1], out[2], out[3]);
    if (tid == 0) xscale[token] = scale;
}

#define LDA 80
#define TILE_B (128 * LDA)

__device__ __forceinline__ void stage_tile(const int8_t* Asrc, const int8_t* Bsrc,
                                           int8_t* dA, int8_t* dB) {
    cp16_g2l(Asrc, dA);
    cp16_g2l(Asrc + 16, dA + 16);
    cp16_g2l(Bsrc, dB);
    cp16_g2l(Bsrc + 16, dB + 16);
}

__device__ __forceinline__ void compute_tile(const int8_t* la, const int8_t* lb,
                                             v8i (&acc)[8]) {
    int4 c0 = *(const int4*)(la);
    int4 c1 = *(const int4*)(la + 32);
    v8i afrag = {c0.x, c0.y, c0.z, c0.w, c1.x, c1.y, c1.z, c1.w};
#pragma unroll
    for (int j = 0; j < 8; ++j) {
        const int8_t* pb = lb + j * 16 * LDA;
        int4 d0 = *(const int4*)(pb);
        int4 d1 = *(const int4*)(pb + 32);
        v8i bfrag = {d0.x, d0.y, d0.z, d0.w, d1.x, d1.y, d1.z, d1.w};
        acc[j] = __builtin_amdgcn_wmma_i32_16x16x64_iu8(
            true, afrag, true, bfrag, acc[j], false, false);
    }
}

__global__ __launch_bounds__(256) void bitgemm_kernel(const int8_t* __restrict__ xq,
                                                      const int8_t* __restrict__ wq,
                                                      const float* __restrict__ xscale,
                                                      const float* __restrict__ bias,
                                                      float* __restrict__ out) {
    __shared__ alignas(16) int8_t lds[4 * TILE_B];

    const int tid  = threadIdx.x;
    const int lane = tid & 31;
    const int wave = tid >> 5;
    const int hi   = lane >> 4;
    const int l16  = lane & 15;

    const int nBase = blockIdx.x * 128;
    const int mBase = blockIdx.y * 128;

    const int row = tid >> 1;
    const int off = (tid & 1) * 32;
    const int stOff = row * LDA + off;

    const int8_t* Ag = xq + (size_t)(mBase + row) * D_IN + off;
    const int8_t* Bg = wq + (size_t)(nBase + row) * D_IN + off;

    int8_t* stA0 = lds + stOff;
    int8_t* stA1 = lds + TILE_B + stOff;
    int8_t* stB0 = lds + 2 * TILE_B + stOff;
    int8_t* stB1 = lds + 3 * TILE_B + stOff;

    const int aFragOff = (wave * 16 + l16) * LDA + hi * 16;
    const int bFragOff = l16 * LDA + hi * 16;
    const int8_t* laBase = lds + aFragOff;
    const int8_t* lbBase = lds + 2 * TILE_B + bFragOff;

    v8i acc[8];
#pragma unroll
    for (int j = 0; j < 8; ++j) acc[j] = (v8i){0, 0, 0, 0, 0, 0, 0, 0};

    stage_tile(Ag, Bg, stA0, stB0);

    for (int k0 = 0; k0 < D_IN; k0 += 128) {
        __syncthreads();
        stage_tile(Ag + k0 + 64, Bg + k0 + 64, stA1, stB1);
        wait_async<4>();
        __syncthreads();
        compute_tile(laBase, lbBase, acc);

        const int kn = (k0 + 128) & (D_IN - 1);
        __syncthreads();
        stage_tile(Ag + kn, Bg + kn, stA0, stB0);
        wait_async<4>();
        __syncthreads();
        compute_tile(laBase + TILE_B, lbBase + TILE_B, acc);
    }

    const int rowBase = mBase + wave * 16 + hi * 8;
    float sc[8];
#pragma unroll
    for (int r = 0; r < 8; ++r) sc[r] = xscale[rowBase + r];

#pragma unroll
    for (int j = 0; j < 8; ++j) {
        const int col = nBase + j * 16 + l16;
        const float bj = bias[col];
#pragma unroll
        for (int r = 0; r < 8; ++r) {
            float val = sc[r] * (float)acc[j][r] + bj;
            __builtin_nontemporal_store(val, &out[(size_t)(rowBase + r) * D_OUT + col]);
        }
    }
}

extern "C" void kernel_launch(void* const* d_in, const int* in_sizes, int n_in,
                              void* d_out, int out_size, void* d_ws, size_t ws_size,
                              hipStream_t stream) {
    const float* x    = (const float*)d_in[0];
    const float* w    = (const float*)d_in[1];
    const float* bias = (const float*)d_in[2];
    float* out        = (float*)d_out;

    uint8_t* ws = (uint8_t*)d_ws;
    float*  sum_abs = (float*)ws;
    int8_t* wq      = (int8_t*)(ws + 256);
    int8_t* xqv     = wq + W_ELEMS;
    float*  xscale  = (float*)(xqv + X_ELEMS);

    init_ws_kernel<<<1, 1, 0, stream>>>(sum_abs);
    wabs_sum_kernel<<<2048, 256, 0, stream>>>(w, sum_abs);
    wquant_kernel<<<8192, 256, 0, stream>>>(w, sum_abs, wq);
    xquant_kernel<<<M_TOT, 256, 0, stream>>>(x, xqv, xscale);

    dim3 grid(D_OUT / 128, M_TOT / 128);
    bitgemm_kernel<<<grid, 256, 0, stream>>>(xqv, wq, xscale, bias, out);
}
